// MoiraiMoEBlock_14516989460791
// MI455X (gfx1250) — compile-verified, matrix operand order checked by reading, not run on hardware
//
#include <hip/hip_runtime.h>
#include <stdint.h>

#define Bt 32768
#define Dd 768
#define Hh 3072
#define Ee 8
#define MT 32          // tokens per tile
#define HC 128         // H chunk (first-GEMM N, second-GEMM K)
#define D2 (Dd/2)      // packed bf16 pairs per x / W1pk row
#define H2 (Hh/2)      // packed bf16 pairs per W2pk row

typedef __attribute__((ext_vector_type(16))) __bf16 v16bf;
typedef __attribute__((ext_vector_type(8)))  float  v8f;
typedef __attribute__((ext_vector_type(4)))  uint32_t u32x4;

union Frag { uint32_t u[8]; v16bf v; };

__device__ __forceinline__ uint32_t pack_bf16(float a, float b) {
    union { float f; uint32_t u; } x, y;
    x.f = a; y.f = b;
    uint32_t lo = (x.u + 0x7FFFu + ((x.u >> 16) & 1u)) >> 16;
    uint32_t hi = (y.u + 0x7FFFu + ((y.u >> 16) & 1u)) >> 16;
    return (hi << 16) | (lo & 0xFFFFu);
}

// K index inside a 16x32 bf16 A/B fragment: VGPR j holds K pair (k, k+1).
// Lanes 0-15: K in {0..7, 16..23}; lanes 16-31: K in {8..15, 24..31}.
__device__ __forceinline__ int kidx(int j, int khalf) {
    return ((j >> 2) << 4) + khalf + ((j & 3) << 1);
}

// A fragment from LDS holding packed-bf16 rows (ldu uint32 per row)
__device__ __forceinline__ v16bf loadA_pk(const uint32_t* base, int m, int ldu,
                                          int kb, int khalf) {
    Frag f;
#pragma unroll
    for (int j = 0; j < 8; ++j)
        f.u[j] = base[m * ldu + ((kb + kidx(j, khalf)) >> 1)];
    return f.v;
}

// B fragment straight from a pre-packed global row (row = one N column,
// packed bf16 pairs along K): lowers to 2x global_load_b128 per lane.
__device__ __forceinline__ v16bf loadB_row(const uint32_t* __restrict__ row,
                                           int kb, int khalf) {
    Frag f;
#pragma unroll
    for (int j = 0; j < 8; ++j)
        f.u[j] = row[(kb + kidx(j, khalf)) >> 1];
    return f.v;
}

__global__ void zero_kernel(float* __restrict__ y, int* __restrict__ counts, long n) {
    long i = (long)blockIdx.x * blockDim.x + threadIdx.x;
    long stride = (long)gridDim.x * blockDim.x;
    for (long p = i; p < n; p += stride) y[p] = 0.f;
    if (i < Ee) counts[i] = 0;
}

// xpk[p] = pack(x[2p], x[2p+1]) : row-major bf16-pair copy of x
__global__ void pack_x_kernel(const float* __restrict__ x,
                              uint32_t* __restrict__ xpk, long npairs) {
    long i = (long)blockIdx.x * blockDim.x + threadIdx.x;
    long stride = (long)gridDim.x * blockDim.x;
    for (long p = i; p < npairs; p += stride)
        xpk[p] = pack_bf16(x[2 * p], x[2 * p + 1]);
}

// in : [Ee][R][C] f32   out: [Ee][C][R/2] packed bf16 pairs along R
// (LDS-tiled so both global sides are coalesced)
__global__ __launch_bounds__(256) void pack_transpose_kernel(
    const float* __restrict__ in, uint32_t* __restrict__ out, int R, int C) {
    __shared__ float t[64][65];
    int e = blockIdx.y;
    int tilesC = C >> 6;
    int tr = blockIdx.x / tilesC;
    int tc = blockIdx.x - tr * tilesC;
    int r0 = tr << 6, c0 = tc << 6;
    const float*  ine  = in  + (size_t)e * R * C;
    uint32_t*     oute = out + (size_t)e * C * (R >> 1);

    int tx = threadIdx.x & 63, ty = threadIdx.x >> 6;   // ty in 0..3
#pragma unroll
    for (int i = 0; i < 16; ++i) {
        int r = (i << 2) + ty;
        t[r][tx] = ine[(size_t)(r0 + r) * C + c0 + tx];
    }
    __syncthreads();
    int dp = threadIdx.x & 31;      // packed-pair index within tile (0..31)
    int nl = threadIdx.x >> 5;      // 0..7
#pragma unroll
    for (int i = 0; i < 8; ++i) {
        int n = (i << 3) + nl;      // column within tile (0..63)
        uint32_t v = pack_bf16(t[2 * dp][n], t[2 * dp + 1][n]);
        oute[(size_t)(c0 + n) * (R >> 1) + (r0 >> 1) + dp] = v;
    }
}

__global__ __launch_bounds__(256) void gate_kernel(
    const float* __restrict__ x, const float* __restrict__ Wg,
    const float* __restrict__ bg, int* __restrict__ counts,
    int* __restrict__ bidx, float* __restrict__ bw) {
    int lane = threadIdx.x & 31;
    int wv   = threadIdx.x >> 5;
    int t    = blockIdx.x * 8 + wv;

    float acc[Ee];
#pragma unroll
    for (int e = 0; e < Ee; ++e) acc[e] = 0.f;

    const float* xr = x + (size_t)t * Dd;
    for (int d = lane; d < Dd; d += 32) {
        float xv = xr[d];
        const float* wr = Wg + d * Ee;
#pragma unroll
        for (int e = 0; e < Ee; ++e) acc[e] += xv * wr[e];
    }
#pragma unroll
    for (int e = 0; e < Ee; ++e) {
#pragma unroll
        for (int off = 16; off > 0; off >>= 1)
            acc[e] += __shfl_xor(acc[e], off, 32);
    }
    if (lane == 0) {
#pragma unroll
        for (int e = 0; e < Ee; ++e) acc[e] += bg[e];
        int e0 = 0; float v0 = acc[0];
#pragma unroll
        for (int e = 1; e < Ee; ++e) if (acc[e] > v0) { v0 = acc[e]; e0 = e; }
        int e1 = -1; float v1 = -3.4e38f;
#pragma unroll
        for (int e = 0; e < Ee; ++e)
            if (e != e0 && acc[e] > v1) { v1 = acc[e]; e1 = e; }
        float ex = __expf(v1 - v0);
        float p0 = 1.f / (1.f + ex);
        float p1 = ex / (1.f + ex);
        int pos0 = atomicAdd(&counts[e0], 1);
        bidx[e0 * Bt + pos0] = t; bw[e0 * Bt + pos0] = p0;
        int pos1 = atomicAdd(&counts[e1], 1);
        bidx[e1 * Bt + pos1] = t; bw[e1 * Bt + pos1] = p1;
    }
}

__global__ __launch_bounds__(256) void ffn_kernel(
    const uint32_t* __restrict__ xpk,    // [Bt][D2] packed bf16
    const uint32_t* __restrict__ W1pk,   // [Ee][Hh][D2]  (col-major, K-packed)
    const float* __restrict__ b1,
    const uint32_t* __restrict__ W2pk,   // [Ee][Dd][H2]  (col-major, K-packed)
    const float* __restrict__ b2,
    const int* __restrict__ counts, const int* __restrict__ bidx,
    const float* __restrict__ bw, float* __restrict__ y) {
    __shared__ uint32_t xs[MT * D2];        // 48 KB: x tile, packed bf16
    __shared__ uint32_t hs_pk[MT * (HC/2)]; // 8 KB : h chunk, packed bf16

    int e  = blockIdx.y;
    int count = counts[e];
    int m0 = blockIdx.x * MT;
    if (m0 >= count) return;

    const uint32_t* W1e = W1pk + (size_t)e * Hh * D2;
    const uint32_t* W2e = W2pk + (size_t)e * Dd * H2;
    const float* b1e = b1 + (size_t)e * Hh;
    const float* b2e = b2 + (size_t)e * Dd;
    const int*   tide = bidx + (size_t)e * Bt;
    const float* gwe  = bw  + (size_t)e * Bt;

    int tid  = threadIdx.x;
    int lane = tid & 31;
    int w    = tid >> 5;          // wave id 0..7
    int lm   = lane & 15;
    int khalf = (lane >> 4) * 8;

    // ---- stage x tile: pure b128 global -> LDS copies ----
    for (int q = tid; q < MT * (D2 / 4); q += 256) {
        int row = q / (D2 / 4), c = (q - row * (D2 / 4)) * 4;
        int gm  = m0 + row;
        int tok = (gm < count) ? tide[gm] : 0;
        *(u32x4*)(xs + row * D2 + c) =
            *(const u32x4*)(xpk + (size_t)tok * D2 + c);
    }
    __syncthreads();

    const v8f zero8 = {0.f, 0.f, 0.f, 0.f, 0.f, 0.f, 0.f, 0.f};
    v8f acc[12];
#pragma unroll
    for (int i = 0; i < 12; ++i) acc[i] = zero8;

    for (int hc = 0; hc < Hh / HC; ++hc) {
        int c0 = hc * HC;
        // ---- first GEMM: h[32 x 128] = x[32 x 768] * W1[768 x 128] ----
        // B fragments straight from W1pk row (one row per N column).
        const uint32_t* w1row = W1e + (size_t)(c0 + w * 16 + lm) * D2;
        v8f hacc0 = zero8, hacc1 = zero8;
        for (int k = 0; k < Dd / 32; ++k) {
            int d0 = k * 32;
            v16bf a0 = loadA_pk(xs, lm, D2, d0, khalf);
            v16bf a1 = loadA_pk(xs, 16 + lm, D2, d0, khalf);
            v16bf bf = loadB_row(w1row, d0, khalf);
            if (k + 1 < Dd / 32)
                __builtin_prefetch(w1row + (d0 >> 1) + 16, 0, 1);
            hacc0 = __builtin_amdgcn_wmma_f32_16x16x32_bf16(
                false, a0, false, bf, (short)0, hacc0, false, false);
            hacc1 = __builtin_amdgcn_wmma_f32_16x16x32_bf16(
                false, a1, false, bf, (short)0, hacc1, false, false);
        }
        __syncthreads();   // all waves done reading hs_pk of previous chunk
        // ---- bias + relu, lane-pair pack to bf16, store to hs_pk ----
        {
            int n = w * 16 + lm;               // column within chunk (0..127)
            float b1v = b1e[c0 + n];
#pragma unroll
            for (int m2 = 0; m2 < 2; ++m2) {
#pragma unroll
                for (int r = 0; r < 8; ++r) {
                    int m = m2 * 16 + r + ((lane >> 4) << 3);
                    float v = (m2 ? hacc1[r] : hacc0[r]) + b1v;
                    v = v > 0.f ? v : 0.f;
                    float vo = __shfl_xor(v, 1);
                    if ((lm & 1) == 0)
                        hs_pk[m * (HC / 2) + (n >> 1)] = pack_bf16(v, vo);
                }
            }
        }
        __syncthreads();
        // ---- second GEMM: out[32 x 768] += h[32 x 128] * W2[128 x 768] ----
#pragma unroll
        for (int ks = 0; ks < HC / 32; ++ks) {
            v16bf af0 = loadA_pk(hs_pk, lm,      HC / 2, ks * 32, khalf);
            v16bf af1 = loadA_pk(hs_pk, 16 + lm, HC / 2, ks * 32, khalf);
            int kb = c0 + ks * 32;
#pragma unroll
            for (int j = 0; j < 6; ++j) {
                const uint32_t* w2row =
                    W2e + (size_t)(w * 96 + j * 16 + lm) * H2;
                v16bf bf2 = loadB_row(w2row, kb, khalf);
                acc[j * 2 + 0] = __builtin_amdgcn_wmma_f32_16x16x32_bf16(
                    false, af0, false, bf2, (short)0, acc[j * 2 + 0],
                    false, false);
                acc[j * 2 + 1] = __builtin_amdgcn_wmma_f32_16x16x32_bf16(
                    false, af1, false, bf2, (short)0, acc[j * 2 + 1],
                    false, false);
            }
        }
    }

    // ---- scatter: y[token] += gate_w * (out + b2) ----
#pragma unroll
    for (int j = 0; j < 6; ++j) {
        int n = w * 96 + j * 16 + lm;
        float b2v = b2e[n];
#pragma unroll
        for (int m2 = 0; m2 < 2; ++m2) {
#pragma unroll
            for (int r = 0; r < 8; ++r) {
                int m  = m2 * 16 + r + ((lane >> 4) << 3);
                int gm = m0 + m;
                float gwt = (gm < count) ? gwe[gm] : 0.f;
                int   tok = (gm < count) ? tide[gm] : 0;
                atomicAdd(y + (size_t)tok * Dd + n,
                          gwt * (acc[j * 2 + m2][r] + b2v));
            }
        }
    }
}

extern "C" void kernel_launch(void* const* d_in, const int* in_sizes, int n_in,
                              void* d_out, int out_size, void* d_ws, size_t ws_size,
                              hipStream_t stream) {
    const float* x  = (const float*)d_in[0];
    const float* Wg = (const float*)d_in[1];
    const float* bg = (const float*)d_in[2];
    const float* W1 = (const float*)d_in[3];
    const float* b1 = (const float*)d_in[4];
    const float* W2 = (const float*)d_in[5];
    const float* b2 = (const float*)d_in[6];
    float* y = (float*)d_out;

    char* ws = (char*)d_ws;
    int*      counts = (int*)ws;                          // 32 B
    int*      bidx   = (int*)(ws + 1024);                 // 1 MB
    float*    bw     = (float*)(ws + 1024 + 4 * Ee * Bt); // 1 MB
    uint32_t* xpk    = (uint32_t*)(ws + ((size_t)4  << 20)); // 48 MB
    uint32_t* W1pk   = (uint32_t*)(ws + ((size_t)52 << 20)); // 36 MB
    uint32_t* W2pk   = (uint32_t*)(ws + ((size_t)88 << 20)); // 36 MB (end 124 MB)

    zero_kernel<<<2048, 256, 0, stream>>>(y, counts, (long)Bt * Dd);
    pack_x_kernel<<<2048, 256, 0, stream>>>(x, xpk, (long)Bt * D2);
    pack_transpose_kernel<<<dim3((Dd / 64) * (Hh / 64), Ee), 256, 0, stream>>>(
        W1, W1pk, Dd, Hh);
    pack_transpose_kernel<<<dim3((Hh / 64) * (Dd / 64), Ee), 256, 0, stream>>>(
        W2, W2pk, Hh, Dd);
    gate_kernel<<<Bt / 8, 256, 0, stream>>>(x, Wg, bg, counts, bidx, bw);
    ffn_kernel<<<dim3(Bt / MT, Ee), 256, 0, stream>>>(
        xpk, W1pk, b1, W2pk, b2, counts, bidx, bw, y);
}
